// MiddleLayerDecoder_38044820308123
// MI455X (gfx1250) — matrix operand order repaired and checked by reading, not compiled, not run on hardware
//
#include <hip/hip_runtime.h>
#include <cstdint>
#include <cstddef>

// ---------------------------------------------------------------------------
// Problem constants (match reference)
// ---------------------------------------------------------------------------
#define NNODES 50000
#define NNODES_PAD 50048          // gridY*128 : lets GEMM epilogue skip row checks
#define KPTS   8
#define NPTS   (NNODES * KPTS)    // 400000

// ---------------------------------------------------------------------------
// CDNA5 WMMA types / helpers (wave32, 16x16x32 bf16 -> f32)
// ---------------------------------------------------------------------------
typedef __attribute__((ext_vector_type(16))) __bf16 v16bf;
typedef __attribute__((ext_vector_type(8)))  __bf16 v8bf;
typedef __attribute__((ext_vector_type(8)))  float  v8f;

__device__ __forceinline__ v16bf cat16(v8bf lo, v8bf hi) {
  return __builtin_shufflevector(lo, hi, 0,1,2,3,4,5,6,7,8,9,10,11,12,13,14,15);
}

__device__ __forceinline__ v8f wmma_bf16(v16bf a, v16bf b, v8f c) {
  // 8 args: (neg_a, A, neg_b, B, c_mod, C, reuse_a, reuse_b)
  return __builtin_amdgcn_wmma_f32_16x16x32_bf16(false, a, false, b, (short)0, c,
                                                 false, false);
}

// ---------------------------------------------------------------------------
// Small prep kernels
// ---------------------------------------------------------------------------
// Wt[o][i] = bf16(W[i][o])  (weights stored transposed so a WMMA B-fragment is
// a contiguous 32B run per lane)
__global__ void transpose_cast_kernel(const float* __restrict__ W,
                                      __bf16* __restrict__ Wt,
                                      int din, int dout) {
  int idx = blockIdx.x * blockDim.x + threadIdx.x;
  int total = din * dout;
  if (idx >= total) return;
  int o = idx / din;          // output channel (slow)  -> Wt row
  int i = idx - o * din;      // input channel  (fast)  -> contiguous writes
  Wt[(size_t)o * din + i] = (__bf16)W[(size_t)i * dout + o];
}

// cat[:, 0:256] = bf16(input_features); cat has ld = 320 (cols 256..319 will be
// filled with feats by the third node-MLP GEMM). 8 elements / thread.
__global__ void cast_input_kernel(const float* __restrict__ in,
                                  __bf16* __restrict__ cat) {
  int t = blockIdx.x * blockDim.x + threadIdx.x;   // N*32 threads
  if (t >= NNODES * 32) return;
  int r = t >> 5;
  int c = (t & 31) * 8;
  const float4* s = (const float4*)(in + (size_t)r * 256 + c);
  float4 a = s[0], b = s[1];
  v8bf h;
  h[0] = (__bf16)a.x; h[1] = (__bf16)a.y; h[2] = (__bf16)a.z; h[3] = (__bf16)a.w;
  h[4] = (__bf16)b.x; h[5] = (__bf16)b.y; h[6] = (__bf16)b.z; h[7] = (__bf16)b.w;
  *(v8bf*)(cat + (size_t)r * 320 + c) = h;
}

__global__ void cluster_kernel(int* __restrict__ out) {
  int idx = blockIdx.x * blockDim.x + threadIdx.x;
  if (idx < NPTS) out[idx] = idx >> 3;   // repeat(arange(N), K)
}

// ---------------------------------------------------------------------------
// Generic WMMA GEMM:  out = act(A[M,K](bf16) @ Wt[Nout,K]^T + bias)
// Block: 256 threads (8 wave32 in 4x2), block tile 128x128, K-step 32.
// REQUIRES:
//   - K % 32 == 0 (all call sites), so staging needs no K-bound check.
//   - Output buffer has capacity for gridY*128 rows (callers pad / the one
//     d_out writer is overwritten later), so epilogue needs no row check.
// Source rows are clamped: garbage rows/cols cannot pollute in-range outputs.
// ---------------------------------------------------------------------------
#define TLD 40   // 32 + 8 bf16 pad (80B row stride: 16B aligned, conflict-light)

template <bool RELU, bool OUT_BF16>
__global__ __launch_bounds__(256)
void gemm_bias_act(const __bf16* __restrict__ A, int lda,
                   const __bf16* __restrict__ Bt, int ldb,
                   const float* __restrict__ bias,
                   int M, int Nout, int K,
                   float* __restrict__ outF, __bf16* __restrict__ outH,
                   int ldo, int colOff) {
  __shared__ __bf16 ldsA[128 * TLD];
  __shared__ __bf16 ldsB[128 * TLD];

  const int tid  = threadIdx.x;
  const int lane = tid & 31;
  const int wave = tid >> 5;
  const int wr = wave >> 1;                 // 0..3 : 32-row strip
  const int wc = wave & 1;                  // 0..1 : 64-col strip
  const int mBase = blockIdx.y * 128;
  const int nBase = blockIdx.x * 128;

  v8f acc[2][4];
#pragma unroll
  for (int mt = 0; mt < 2; ++mt)
#pragma unroll
    for (int nt = 0; nt < 4; ++nt)
#pragma unroll
      for (int e = 0; e < 8; ++e) acc[mt][nt][e] = 0.0f;

  const int srow = tid >> 1;                // 0..127
  const int scol = (tid & 1) * 16;          // 0 or 16

  // clamped source rows: unconditional, divergence-free staging
  const int gA = min(mBase + srow, M - 1);
  const int gB = min(nBase + srow, Nout - 1);
  const __bf16* srcA = A  + (size_t)gA * lda + scol;
  const __bf16* srcB = Bt + (size_t)gB * ldb + scol;
  uint4* dstA = (uint4*)&ldsA[srow * TLD + scol];
  uint4* dstB = (uint4*)&ldsB[srow * TLD + scol];

  for (int kk = 0; kk < K; kk += 32) {
    // ---- stage A/B tiles [128 x 32], two 16B copies each ----
    {
      const uint4* sa = (const uint4*)(srcA + kk);
      const uint4* sb = (const uint4*)(srcB + kk);
      uint4 a0 = sa[0], a1 = sa[1];
      uint4 b0 = sb[0], b1 = sb[1];
      dstA[0] = a0; dstA[1] = a1;
      dstB[0] = b0; dstB[1] = b1;
    }
    __syncthreads();

    // ---- fragments + 8 WMMA ----
    const int l16 = lane & 15;
    const int ksA = (lane >> 4) * 16;       // A: k in {ks..ks+15}
    const int ksB = (lane >> 4) * 16;       // B: k in {ks2..ks2+15}
    v16bf afrag[2], bfrag[4];
#pragma unroll
    for (int mt = 0; mt < 2; ++mt) {
      const __bf16* p = &ldsA[(wr * 32 + mt * 16 + l16) * TLD];
      v8bf lo = *(const v8bf*)(p + ksA);
      v8bf hi = *(const v8bf*)(p + ksA + 8);
      afrag[mt] = cat16(lo, hi);
    }
#pragma unroll
    for (int nt = 0; nt < 4; ++nt) {
      const __bf16* p = &ldsB[(wc * 64 + nt * 16 + l16) * TLD + ksB];
      v8bf lo = *(const v8bf*)(p);
      v8bf hi = *(const v8bf*)(p + 8);
      bfrag[nt] = cat16(lo, hi);
    }
#pragma unroll
    for (int mt = 0; mt < 2; ++mt)
#pragma unroll
      for (int nt = 0; nt < 4; ++nt)
        acc[mt][nt] = wmma_bf16(afrag[mt], bfrag[nt], acc[mt][nt]);
    __syncthreads();
  }

  // ---- epilogue: C layout vgpr r -> m = 8*(lane>=16)+r, n = lane&15 ----
  // No row bound (buffers padded / later overwritten); col bound hoisted.
  const int l16 = lane & 15;
  const int mh  = (lane >> 4) * 8;
#pragma unroll
  for (int mt = 0; mt < 2; ++mt) {
#pragma unroll
    for (int nt = 0; nt < 4; ++nt) {
      int col = nBase + wc * 64 + nt * 16 + l16;
      if (col < Nout) {
        float bv = bias[col];
#pragma unroll
        for (int r = 0; r < 8; ++r) {
          int row = mBase + wr * 32 + mt * 16 + mh + r;
          float v = acc[mt][nt][r] + bv;
          if (RELU) v = fmaxf(v, 0.0f);
          size_t o = (size_t)row * ldo + colOff + col;
          if (OUT_BF16) outH[o] = (__bf16)v;
          else          outF[o] = v;
        }
      }
    }
  }
}

// ---------------------------------------------------------------------------
// Fused per-point kernel. Block = 128 points (16 nodes), 256 threads.
// Phase 0: d0 = relu(shared[node] + rel . W0[320:323])         -> LDS (bf16)
// Phase 1: d1 = relu(d0 @ W1 + b1)  (A from LDS, B from L2)    -> LDS (bf16)
// Phase 2: out = relu(d1 @ W2 + b2)                            -> HBM (f32)
// Dynamic LDS: 2 * 128 * 264 bf16 = 132 KB (gfx1250: 320 KB/WGP).
// ---------------------------------------------------------------------------
#define DLD 264   // 256 + 8 pad

__global__ __launch_bounds__(256)
void decode_fused_kernel(const float* __restrict__ shared_f,   // [N,256] = cat@W0a + b0
                         const float* __restrict__ rel,        // [N*K,3]
                         const float* __restrict__ w0b,        // [3,256] (= W0 rows 320..322)
                         const __bf16* __restrict__ w1t,       // [256,256] transposed
                         const float* __restrict__ b1,
                         const __bf16* __restrict__ w2t,       // [256,256] transposed
                         const float* __restrict__ b2,
                         float* __restrict__ out) {             // [N*K,256]
  extern __shared__ __bf16 smem[];
  __bf16* d0 = smem;                 // [128][DLD]
  __bf16* d1 = smem + 128 * DLD;     // [128][DLD]

  const int tid  = threadIdx.x;
  const int lane = tid & 31;
  const int wave = tid >> 5;
  const int wr = wave >> 1;          // 0..3
  const int wc = wave & 1;           // 0..1
  const int pb = blockIdx.x;         // point-tile index (3125 tiles, exact)

  // ---- phase 0: expand layer-0 preactivation (shared part + rel part) ----
  {
    int r  = tid >> 1;               // 0..127 local point
    int c0 = (tid & 1) * 128;
    int p    = pb * 128 + r;
    int node = pb * 16 + (r >> 3);
    float rx = rel[p * 3 + 0];
    float ry = rel[p * 3 + 1];
    float rz = rel[p * 3 + 2];
    const float* sh = shared_f + (size_t)node * 256;
    for (int c = c0; c < c0 + 128; c += 8) {
      float4 s0 = *(const float4*)(sh + c);
      float4 s1 = *(const float4*)(sh + c + 4);
      float4 x0 = *(const float4*)(w0b + c);
      float4 x1 = *(const float4*)(w0b + c + 4);
      float4 y0 = *(const float4*)(w0b + 256 + c);
      float4 y1 = *(const float4*)(w0b + 256 + c + 4);
      float4 z0 = *(const float4*)(w0b + 512 + c);
      float4 z1 = *(const float4*)(w0b + 512 + c + 4);
      float v0 = s0.x + rx * x0.x + ry * y0.x + rz * z0.x;
      float v1 = s0.y + rx * x0.y + ry * y0.y + rz * z0.y;
      float v2 = s0.z + rx * x0.z + ry * y0.z + rz * z0.z;
      float v3 = s0.w + rx * x0.w + ry * y0.w + rz * z0.w;
      float v4 = s1.x + rx * x1.x + ry * y1.x + rz * z1.x;
      float v5 = s1.y + rx * x1.y + ry * y1.y + rz * z1.y;
      float v6 = s1.z + rx * x1.z + ry * y1.z + rz * z1.z;
      float v7 = s1.w + rx * x1.w + ry * y1.w + rz * z1.w;
      v8bf h;
      h[0] = (__bf16)fmaxf(v0, 0.0f); h[1] = (__bf16)fmaxf(v1, 0.0f);
      h[2] = (__bf16)fmaxf(v2, 0.0f); h[3] = (__bf16)fmaxf(v3, 0.0f);
      h[4] = (__bf16)fmaxf(v4, 0.0f); h[5] = (__bf16)fmaxf(v5, 0.0f);
      h[6] = (__bf16)fmaxf(v6, 0.0f); h[7] = (__bf16)fmaxf(v7, 0.0f);
      *(v8bf*)(&d0[r * DLD + c]) = h;
    }
  }
  __syncthreads();

  const int l16 = lane & 15;
  const int ksA = (lane >> 4) * 16;
  const int ksB = (lane >> 4) * 16;
  const int mh  = (lane >> 4) * 8;

  // ---- phase 1: d1 = relu(d0 @ W1 + b1) ----
#pragma unroll
  for (int nb = 0; nb < 2; ++nb) {
    v8f acc[2][4];
#pragma unroll
    for (int mt = 0; mt < 2; ++mt)
#pragma unroll
      for (int nt = 0; nt < 4; ++nt)
#pragma unroll
        for (int e = 0; e < 8; ++e) acc[mt][nt][e] = 0.0f;

    for (int kk = 0; kk < 256; kk += 32) {
      v16bf af[2];
#pragma unroll
      for (int mt = 0; mt < 2; ++mt) {
        const __bf16* p = &d0[(wr * 32 + mt * 16 + l16) * DLD + kk];
        af[mt] = cat16(*(const v8bf*)(p + ksA), *(const v8bf*)(p + ksA + 8));
      }
#pragma unroll
      for (int nt = 0; nt < 4; ++nt) {
        int n = nb * 128 + wc * 64 + nt * 16 + l16;
        const __bf16* bp = w1t + (size_t)n * 256 + kk + ksB;
        __builtin_prefetch(bp + 32, 0, 1);            // global_prefetch_b8
        v16bf bf_ = cat16(*(const v8bf*)bp, *(const v8bf*)(bp + 8));
#pragma unroll
        for (int mt = 0; mt < 2; ++mt)
          acc[mt][nt] = wmma_bf16(af[mt], bf_, acc[mt][nt]);
      }
    }
#pragma unroll
    for (int mt = 0; mt < 2; ++mt)
#pragma unroll
      for (int nt = 0; nt < 4; ++nt) {
        int col = nb * 128 + wc * 64 + nt * 16 + l16;
        float bv = b1[col];
#pragma unroll
        for (int r = 0; r < 8; ++r) {
          int row = wr * 32 + mt * 16 + mh + r;
          d1[row * DLD + col] = (__bf16)fmaxf(acc[mt][nt][r] + bv, 0.0f);
        }
      }
  }
  __syncthreads();   // cross-wave: phase 2 reads all columns of d1

  // ---- phase 2: out = relu(d1 @ W2 + b2) ----
#pragma unroll
  for (int nb = 0; nb < 2; ++nb) {
    v8f acc[2][4];
#pragma unroll
    for (int mt = 0; mt < 2; ++mt)
#pragma unroll
      for (int nt = 0; nt < 4; ++nt)
#pragma unroll
        for (int e = 0; e < 8; ++e) acc[mt][nt][e] = 0.0f;

    for (int kk = 0; kk < 256; kk += 32) {
      v16bf af[2];
#pragma unroll
      for (int mt = 0; mt < 2; ++mt) {
        const __bf16* p = &d1[(wr * 32 + mt * 16 + l16) * DLD + kk];
        af[mt] = cat16(*(const v8bf*)(p + ksA), *(const v8bf*)(p + ksA + 8));
      }
#pragma unroll
      for (int nt = 0; nt < 4; ++nt) {
        int n = nb * 128 + wc * 64 + nt * 16 + l16;
        const __bf16* bp = w2t + (size_t)n * 256 + kk + ksB;
        v16bf bf_ = cat16(*(const v8bf*)bp, *(const v8bf*)(bp + 8));
#pragma unroll
        for (int mt = 0; mt < 2; ++mt)
          acc[mt][nt] = wmma_bf16(af[mt], bf_, acc[mt][nt]);
      }
    }
#pragma unroll
    for (int mt = 0; mt < 2; ++mt)
#pragma unroll
      for (int nt = 0; nt < 4; ++nt) {
        int col = nb * 128 + wc * 64 + nt * 16 + l16;
        float bv = b2[col];
#pragma unroll
        for (int r = 0; r < 8; ++r) {
          int row = pb * 128 + wr * 32 + mt * 16 + mh + r;
          out[(size_t)row * 256 + col] = fmaxf(acc[mt][nt][r] + bv, 0.0f);
        }
      }
  }
}

// ---------------------------------------------------------------------------
// Host launcher
// ---------------------------------------------------------------------------
extern "C" void kernel_launch(void* const* d_in, const int* in_sizes, int n_in,
                              void* d_out, int out_size, void* d_ws, size_t ws_size,
                              hipStream_t stream) {
  const float* in_f  = (const float*)d_in[0];
  const float* Wg0   = (const float*)d_in[1];
  const float* bg0   = (const float*)d_in[2];
  const float* Wg1   = (const float*)d_in[3];
  const float* bg1   = (const float*)d_in[4];
  const float* Wg2   = (const float*)d_in[5];
  const float* bg2   = (const float*)d_in[6];
  const float* Wdec  = (const float*)d_in[7];
  const float* bdec  = (const float*)d_in[8];
  const float* W0    = (const float*)d_in[9];
  const float* b0    = (const float*)d_in[10];
  const float* W1    = (const float*)d_in[11];
  const float* b1    = (const float*)d_in[12];
  const float* W2    = (const float*)d_in[13];
  const float* b2    = (const float*)d_in[14];

  float* outF = (float*)d_out;
  const size_t REL_OFF = (size_t)NPTS * 3;              // relative_points
  const size_t DEC_OFF = REL_OFF + (size_t)NPTS * 256;  // decoded, then cluster

  // ---- workspace carve (row dims padded to NNODES_PAD: see GEMM epilogue) --
  char* wsp = (char*)d_ws;
  auto carve = [&](size_t bytes) -> char* {
    char* p = wsp;
    wsp += (bytes + 255) & ~(size_t)255;
    return p;
  };
  __bf16* cat    = (__bf16*)carve((size_t)NNODES_PAD * 320 * 2); // [N,320]=[inp|feats]
  __bf16* h0     = (__bf16*)carve((size_t)NNODES_PAD * 256 * 2);
  __bf16* h1     = (__bf16*)carve((size_t)NNODES_PAD * 128 * 2);
  float*  sharedP= (float*) carve((size_t)NNODES_PAD * 256 * 4); // layer-0 shared + b0
  __bf16* wg0t   = (__bf16*)carve(256 * 256 * 2);
  __bf16* wg1t   = (__bf16*)carve(128 * 256 * 2);
  __bf16* wg2t   = (__bf16*)carve(64 * 128 * 2);
  __bf16* wdect  = (__bf16*)carve(24 * 64 * 2);
  __bf16* w0at   = (__bf16*)carve(256 * 320 * 2);
  __bf16* w1t    = (__bf16*)carve(256 * 256 * 2);
  __bf16* w2t    = (__bf16*)carve(256 * 256 * 2);
  const float* w0b = W0 + 320 * 256;              // last 3 rows of W0, used as-is

  // ---- prep: casts / transposes ----
  {
    int tot = NNODES * 32;
    cast_input_kernel<<<(tot + 255) / 256, 256, 0, stream>>>(in_f, cat);
  }
  auto tc = [&](const float* W, __bf16* Wt, int din, int dout) {
    int tot = din * dout;
    transpose_cast_kernel<<<(tot + 255) / 256, 256, 0, stream>>>(W, Wt, din, dout);
  };
  tc(Wg0, wg0t, 256, 256);
  tc(Wg1, wg1t, 256, 128);
  tc(Wg2, wg2t, 128, 64);
  tc(Wdec, wdect, 64, 24);
  tc(W0, w0at, 320, 256);   // only rows 0..319 of W0
  tc(W1, w1t, 256, 256);
  tc(W2, w2t, 256, 256);

  // ---- node-level MLP (WMMA GEMMs) ----
  // relu + bf16 output
  auto gemmH = [&](const __bf16* A, int lda, const __bf16* Bt, int ldb,
                   const float* bias, int M, int Nout, int K,
                   __bf16* oH, int ldo, int colOff) {
    dim3 grid((Nout + 127) / 128, (M + 127) / 128);
    gemm_bias_act<true, true><<<grid, 256, 0, stream>>>(
        A, lda, Bt, ldb, bias, M, Nout, K, nullptr, oH, ldo, colOff);
  };
  // no relu + f32 output
  auto gemmF = [&](const __bf16* A, int lda, const __bf16* Bt, int ldb,
                   const float* bias, int M, int Nout, int K,
                   float* oF, int ldo) {
    dim3 grid((Nout + 127) / 128, (M + 127) / 128);
    gemm_bias_act<false, false><<<grid, 256, 0, stream>>>(
        A, lda, Bt, ldb, bias, M, Nout, K, oF, nullptr, ldo, 0);
  };
  // h0 = relu(inp @ Wg0 + bg0)          [N,256]
  gemmH(cat, 320, wg0t, 256, bg0, NNODES, 256, 256, h0, 256, 0);
  // h1 = relu(h0 @ Wg1 + bg1)           [N,128]
  gemmH(h0, 256, wg1t, 256, bg1, NNODES, 128, 256, h1, 128, 0);
  // feats = relu(h1 @ Wg2 + bg2) -> cat[:,256:320]
  gemmH(h1, 128, wg2t, 128, bg2, NNODES, 64, 128, cat, 320, 256);
  // relative_points = feats @ Wdec + bdec -> d_out[0 : N*K*3] (f32, no relu).
  // Rows 50000..50047 spill <=1152 floats into the decoded region, which the
  // fused kernel fully overwrites afterwards on the same stream.
  gemmF(cat + 256, 320, wdect, 64, bdec, NNODES, 24, 64, outF, 24);
  // shared = [inp|feats] @ W0[0:320] + b0   [N,256] f32, no relu
  gemmF(cat, 320, w0at, 320, b0, NNODES, 256, 320, sharedP, 256);

  // ---- fused per-point MLP: expand + 2 WMMA layers in LDS ----
  {
    size_t lds_bytes = (size_t)2 * 128 * DLD * sizeof(__bf16);  // 135168 B
    decode_fused_kernel<<<NPTS / 128, 256, lds_bytes, stream>>>(
        sharedP, outF /*rel*/, w0b, w1t, b1, w2t, b2, outF + REL_OFF);
  }

  // ---- cluster indices (int32 bit-pattern into tail of d_out) ----
  cluster_kernel<<<(NPTS + 255) / 256, 256, 0, stream>>>((int*)(outF + DEC_OFF));
}
